// POSSM_Backbone_GRU_74972949119402
// MI455X (gfx1250) — matrix operand order repaired and checked by reading, not compiled, not run on hardware
//
#include <hip/hip_runtime.h>
#include <hip/hip_bf16.h>

// ---------------------------------------------------------------------------
// GRU (POSSM backbone): B=128, T=512, D=512, H=256, 3H=768
// Phase 1: gx = x @ W_ih^T + b_ih  (bf16 WMMA, f32 accumulate)
// Phase 2: batch-parallel scan, gh = h @ W_hh^T per step (bf16 WMMA)
// ---------------------------------------------------------------------------

typedef __attribute__((ext_vector_type(16))) __bf16 v16bf;
typedef __attribute__((ext_vector_type(8)))  float  v8f;

#define B_   128
#define T_   512
#define D_   512
#define H_   256
#define G3_  768   // 3*H

union BF16x16 { v16bf v; uint4 q[2]; unsigned short u[16]; };

__device__ __forceinline__ unsigned short f2bf(float f) {
  unsigned int u = __float_as_uint(f);
  u += 0x7FFFu + ((u >> 16) & 1u);            // round-to-nearest-even
  return (unsigned short)(u >> 16);
}
__device__ __forceinline__ float bf2f(unsigned short h) {
  return __uint_as_float(((unsigned int)h) << 16);
}
__device__ __forceinline__ float sigf(float x)  { return 1.0f / (1.0f + __expf(-x)); }
__device__ __forceinline__ float tanhf_fast(float x) { return 1.0f - 2.0f / (1.0f + __expf(2.0f * x)); }

// ---------------------------------------------------------------------------
// Kernel 0: convert weights f32 -> bf16
// ---------------------------------------------------------------------------
__global__ void cvt_weights(const float* __restrict__ wih, const float* __restrict__ whh,
                            unsigned short* __restrict__ wbih, unsigned short* __restrict__ wbhh) {
  int i = blockIdx.x * blockDim.x + threadIdx.x;
  const int N1 = G3_ * D_;      // 393216
  const int N2 = G3_ * H_;      // 196608
  if (i < N1)            wbih[i]      = f2bf(wih[i]);
  else if (i < N1 + N2)  wbhh[i - N1] = f2bf(whh[i - N1]);
}

// ---------------------------------------------------------------------------
// Kernel 1: gx[t][b][n] = sum_k x[(b,t),k] * W_ih[n,k] + b_ih[n], stored bf16
// grid = 4096 (M-tiles of 16 rows of B*T), block = 256 (8 waves, 6 N-tiles each)
// ---------------------------------------------------------------------------
#define PADX 8
__global__ __launch_bounds__(256) void gemm_gx(
    const float* __restrict__ x,             // (B*T, 512) f32
    const unsigned short* __restrict__ wbih, // (768, 512) bf16
    const float* __restrict__ b_ih,          // (768)
    unsigned short* __restrict__ gx)         // (T, B, 768) bf16
{
  __shared__ __align__(16) unsigned short xs[16][D_ + PADX];
  const int tid   = threadIdx.x;
  const int mtile = blockIdx.x;
  const long rowbase = (long)mtile * 16;

  // stage 16x512 f32 tile -> bf16 LDS (coalesced float4 loads)
  #pragma unroll
  for (int i = 0; i < 8; ++i) {
    int v   = tid + i * 256;                 // float4 index 0..2047
    int row = v >> 7;                        // 128 float4 per row
    int c4  = v & 127;
    const float4 f = reinterpret_cast<const float4*>(x + (rowbase + row) * D_)[c4];
    unsigned short* p = &xs[row][c4 * 4];
    p[0] = f2bf(f.x); p[1] = f2bf(f.y); p[2] = f2bf(f.z); p[3] = f2bf(f.w);
  }
  __syncthreads();

  const int wave  = tid >> 5;
  const int lane  = tid & 31;
  const int m     = lane & 15;
  const int hi    = lane >> 4;               // 0/1 (lane half)
  const int koffA = hi * 8;                  // A: second half-lane shifts K by 8
  const int koffB = hi * 8;                  // B: second half-lane shifts K by 8
  const int nt0   = wave * 6;

  v8f acc[6] = {};
  const unsigned short* bp[6];
  #pragma unroll
  for (int j = 0; j < 6; ++j) {
    int n = (nt0 + j) * 16 + m;
    bp[j] = wbih + (long)n * D_ + koffB;
  }

  for (int kk = 0; kk < 16; ++kk) {          // K = 512 = 16 * 32
    const int kb = kk * 32;
    BF16x16 a;
    a.q[0] = *reinterpret_cast<const uint4*>(&xs[m][kb + koffA]);
    a.q[1] = *reinterpret_cast<const uint4*>(&xs[m][kb + koffA + 16]);
    #pragma unroll
    for (int j = 0; j < 6; ++j) {
      BF16x16 b;
      b.q[0] = *reinterpret_cast<const uint4*>(bp[j] + kb);
      b.q[1] = *reinterpret_cast<const uint4*>(bp[j] + kb + 16);
      acc[j] = __builtin_amdgcn_wmma_f32_16x16x32_bf16(
          false, a.v, false, b.v, (short)0, acc[j], false, false);
    }
  }

  // epilogue: + bias, store bf16 into gx laid out [t][b][n]
  #pragma unroll
  for (int j = 0; j < 6; ++j) {
    const int   n    = (nt0 + j) * 16 + m;
    const float bias = b_ih[n];
    #pragma unroll
    for (int r = 0; r < 8; ++r) {
      const long M  = rowbase + r + koffA;   // row in B*T
      const int  bb = (int)(M >> 9);         // / T_
      const int  tt = (int)(M & (T_ - 1));
      gx[((long)tt * B_ + bb) * G3_ + n] = f2bf(acc[j][r] + bias);
    }
  }
}

// ---------------------------------------------------------------------------
// Kernel 2: sequential GRU scan, parallel over batch.
// grid = 8 blocks (16 batch rows each), block = 512 (16 waves).
// Wave w owns h columns [16w, 16w+16): 3 gh tiles (r,z,n) per step.
// ---------------------------------------------------------------------------
#define PADH 8
__global__ __launch_bounds__(512) void gru_scan(
    const unsigned short* __restrict__ gx,   // (T,B,768) bf16
    const unsigned short* __restrict__ wbhh, // (768,256) bf16
    const float* __restrict__ b_hh,          // (768)
    const float* __restrict__ h_prev,        // (B,256) f32
    const unsigned char* __restrict__ bmask, // (B,T) bool
    float* __restrict__ out)                 // (B,T,256) f32
{
  __shared__ __align__(16) unsigned short hbf[16][H_ + PADH]; // bf16 state (WMMA A)
  __shared__ __align__(16) float          hf [16][H_ + 4];    // f32 state (carry)

  const int tid = threadIdx.x;
  const int b0  = blockIdx.x * 16;

  #pragma unroll
  for (int i = 0; i < 8; ++i) {              // init 16x256 state
    int idx = tid + i * 512;
    int row = idx >> 8, col = idx & 255;
    float v = h_prev[(b0 + row) * H_ + col];
    hf[row][col]  = v;
    hbf[row][col] = f2bf(v);
  }
  __syncthreads();

  const int wave  = tid >> 5;
  const int lane  = tid & 31;
  const int m     = lane & 15;
  const int hi    = lane >> 4;
  const int koffA = hi * 8;
  const int koffB = hi * 8;
  const int col0  = wave * 16;               // owned h columns

  const unsigned short* bp[3];
  float bias[3];
  int   ncol[3];
  #pragma unroll
  for (int g = 0; g < 3; ++g) {              // gates r,z,n
    int n   = g * H_ + col0 + m;
    ncol[g] = n;
    bp[g]   = wbhh + (long)n * H_ + koffB;
    bias[g] = b_hh[n];
  }

  for (int t = 0; t < T_; ++t) {
    v8f acc[3] = {};
    #pragma unroll
    for (int kk = 0; kk < 8; ++kk) {         // K = 256 = 8 * 32
      const int kb = kk * 32;
      BF16x16 a;
      a.q[0] = *reinterpret_cast<const uint4*>(&hbf[m][kb + koffA]);
      a.q[1] = *reinterpret_cast<const uint4*>(&hbf[m][kb + koffA + 16]);
      #pragma unroll
      for (int g = 0; g < 3; ++g) {
        BF16x16 b;
        b.q[0] = *reinterpret_cast<const uint4*>(bp[g] + kb);
        b.q[1] = *reinterpret_cast<const uint4*>(bp[g] + kb + 16);
        acc[g] = __builtin_amdgcn_wmma_f32_16x16x32_bf16(
            false, a.v, false, b.v, (short)0, acc[g], false, false);
      }
    }

    // input-side gates for this step (bf16 -> f32)
    float gxa[3][8];
    const long gbase = ((long)t * B_ + b0) * G3_;
    #pragma unroll
    for (int g = 0; g < 3; ++g)
      #pragma unroll
      for (int r = 0; r < 8; ++r)
        gxa[g][r] = bf2f(gx[gbase + (long)(r + koffA) * G3_ + ncol[g]]);

    __syncthreads();                         // all waves done reading hbf

    const int col = col0 + m;
    #pragma unroll
    for (int r = 0; r < 8; ++r) {
      const int   mrow = r + koffA;
      const int   bb   = b0 + mrow;
      const float rg   = sigf(gxa[0][r] + acc[0][r] + bias[0]);
      const float zg   = sigf(gxa[1][r] + acc[1][r] + bias[1]);
      const float ng   = tanhf_fast(gxa[2][r] + rg * (acc[2][r] + bias[2]));
      const float hold = hf[mrow][col];
      const float hnew = (1.0f - zg) * ng + zg * hold;
      const float mk   = bmask[(long)bb * T_ + t] ? 1.0f : 0.0f;
      const float hc   = mk * hnew + (1.0f - mk) * hold;   // freeze on padding
      out[((long)bb * T_ + t) * H_ + col] = mk * hnew;     // padded outputs zero
      hf[mrow][col]  = hc;
      hbf[mrow][col] = f2bf(hc);
    }
    __syncthreads();                         // updates visible for next step
  }
}

// ---------------------------------------------------------------------------
// Host entry
// ---------------------------------------------------------------------------
extern "C" void kernel_launch(void* const* d_in, const int* in_sizes, int n_in,
                              void* d_out, int out_size, void* d_ws, size_t ws_size,
                              hipStream_t stream) {
  const float*          z_t    = (const float*)d_in[0];          // (B,T,8,64)
  const unsigned char*  bmask  = (const unsigned char*)d_in[1];  // (B,T) bool
  const float*          h_prev = (const float*)d_in[2];          // (1,B,H)
  const float*          W_ih   = (const float*)d_in[3];          // (768,512)
  const float*          W_hh   = (const float*)d_in[4];          // (768,256)
  const float*          b_ih   = (const float*)d_in[5];          // (768)
  const float*          b_hh   = (const float*)d_in[6];          // (768)
  float*                out    = (float*)d_out;                  // (B,T,H)

  // workspace layout: [W_ih bf16][W_hh bf16][gx bf16 (T,B,768)]  (~97 MB)
  unsigned short* wbih = (unsigned short*)d_ws;
  unsigned short* wbhh = wbih + (size_t)G3_ * D_;
  unsigned short* gx   = wbhh + (size_t)G3_ * H_;

  (void)in_sizes; (void)n_in; (void)out_size; (void)ws_size;

  const int cvtN = G3_ * D_ + G3_ * H_;
  cvt_weights<<<(cvtN + 255) / 256, 256, 0, stream>>>(W_ih, W_hh, wbih, wbhh);
  gemm_gx<<<(B_ * T_) / 16, 256, 0, stream>>>(z_t, wbih, b_ih, gx);
  gru_scan<<<B_ / 16, 512, 0, stream>>>(gx, wbhh, b_hh, h_prev, bmask, out);
}
